// PathNN_47218870452454
// MI455X (gfx1250) — compile-verified, matrix operand order checked by reading, not run on hardware
//
#include <hip/hip_runtime.h>
#include <cstdint>

// ---------------- sizes ----------------
#define NNODES 50000
#define DIN    128
#define HIDN   64
#define NEDGES 800000
#define NCLS   10
#define EPSV   1e-5f

typedef __bf16   v16bf __attribute__((ext_vector_type(16)));
typedef float    v8f   __attribute__((ext_vector_type(8)));
typedef uint32_t u32;
typedef uint32_t u32x4 __attribute__((ext_vector_type(4)));
typedef unsigned int v4u __attribute__((ext_vector_type(4)));
typedef int v8i __attribute__((ext_vector_type(8)));
typedef int v4i __attribute__((ext_vector_type(4)));

union AFrag { u32 u[8]; v16bf v; };
union BFrag { u32x4 q[2]; v16bf v; };

__device__ __forceinline__ v8f wmma_bf16(v16bf a, v16bf b, v8f c) {
  // emits v_wmma_f32_16x16x32_bf16
  return __builtin_amdgcn_wmma_f32_16x16x32_bf16(false, a, false, b, (short)0, c, false, false);
}

__device__ __forceinline__ v8f splat8(float x) {
  v8f r = {x, x, x, x, x, x, x, x};
  return r;
}

__device__ __forceinline__ float sigf(float x) { return 1.0f / (1.0f + __expf(-x)); }

// A-matrix 16x32 bf16 lane layout: pair j (elements 2j,2j+1) of half `hf`
// holds K = base(j) + 8*hf, base(j) = (j&3)*2 + (j>>2)*16   (ISA 7.12.2)
__device__ __forceinline__ int a_kbase(int j, int hf) {
  return ((j & 3) << 1) + ((j >> 2) << 4) + (hf << 3);
}

__device__ __forceinline__ void load_a_chunk(const __bf16* row, int kc, int hf, u32* u) {
#pragma unroll
  for (int j = 0; j < 8; ++j) {
    int kk = kc * 32 + a_kbase(j, hf);
    u[j] = *(const u32*)(row + kk);
  }
}

__device__ __forceinline__ v16bf load_b(const __bf16* packed, int nt, int kc, int kChunks, int lane) {
  const u32x4* p = (const u32x4*)(packed + ((size_t)(nt * kChunks + kc) * 32 + lane) * 16);
  BFrag b;
  b.q[0] = p[0];
  b.q[1] = p[1];
  return b.v;
}

// ---- TDM: DMA a contiguous run of n bf16 elements (one 1-row tile) into LDS ----
// D# group layouts per CDNA5 ISA 8.3/8.4; groups 2/3 zero (<=2D tensor).
// This toolchain exposes the 6-arg builtin: (v4u g0, v8i g1, v4i, v4i, v8i, i32 cpol).
__device__ __forceinline__ void tdm_load_lds_1d(uint32_t ldsOff, const void* gptr, uint32_t n) {
  uint64_t ga = (uint64_t)(uintptr_t)gptr;
  v4u g0;
  g0[0] = 1u;                                   // count=1, user desc, no gather
  g0[1] = ldsOff;                               // lds_addr (bytes)
  g0[2] = (uint32_t)ga;                         // global_addr[31:0]
  g0[3] = (uint32_t)(ga >> 32) | (2u << 30);    // global_addr[56:32] | type=2 ("image")
  v8i g1;
  g1[0] = (int)(1u << 16);                      // workgroup_mask=0, data_size=1 (2 bytes)
  g1[1] = (int)((n & 0xFFFFu) << 16);           // tensor_dim0 lo16 (bits 63:48)
  g1[2] = (int)(((n >> 16) & 0xFFFFu) | (1u << 16)); // tensor_dim0 hi16 | tensor_dim1=1
  g1[3] = (int)((n & 0xFFFFu) << 16);           // tensor_dim1 hi=0 | tile_dim0=n
  g1[4] = 1;                                    // tile_dim1=1, tile_dim2=0
  g1[5] = (int)n;                               // tensor_dim0_stride lo32
  g1[6] = 0;                                    // stride0 hi16 | stride1 lo16
  g1[7] = 0;                                    // stride1 hi32
  v4i z4 = {0, 0, 0, 0};
  v8i z8 = {0, 0, 0, 0, 0, 0, 0, 0};
  __builtin_amdgcn_tensor_load_to_lds(g0, g1, z4, z4, z8, 0);
}

// -------- pack W[n,k] (row-major, N x K) into per-lane WMMA B-operand layout --------
__global__ void pack_b_kernel(const float* __restrict__ W, __bf16* __restrict__ dst,
                              int K, int kChunks, int nTiles) {
  int idx = blockIdx.x * blockDim.x + threadIdx.x;
  int total = nTiles * kChunks * 32 * 16;
  if (idx >= total) return;
  int e = idx & 15;
  int lane = (idx >> 4) & 31;
  int rest = idx >> 9;
  int kc = rest % kChunks;
  int nt = rest / kChunks;
  int j = e >> 1, rb = e & 1, hf = lane >> 4;
  int n = nt * 16 + (lane & 15);
  int k = kc * 32 + a_kbase(j, hf) + rb;
  dst[idx] = (__bf16)W[(size_t)n * K + k];
}

__global__ void bias_sum_kernel(const float* __restrict__ bi, const float* __restrict__ bh,
                                float* __restrict__ bs) {
  int i = threadIdx.x;
  if (i < 256) bs[i] = bi[i] + bh[i];
}

__global__ void f32_to_bf16_kernel(const float* __restrict__ src, __bf16* __restrict__ dst, int n) {
  int i = blockIdx.x * blockDim.x + threadIdx.x;
  if (i < n) dst[i] = (__bf16)src[i];
}

// -------- generic bf16 WMMA GEMM: Y[r, 0..63] = A[r, 0..K-1] @ Bpacked + bias --------
__global__ __launch_bounds__(128) void gemm_bf16_kernel(const __bf16* __restrict__ A, int K, int kChunks,
                                                        const __bf16* __restrict__ Bp,
                                                        const float* __restrict__ bias,
                                                        float* __restrict__ Y, int rowTiles) {
  int wave = threadIdx.x >> 5;
  int lane = threadIdx.x & 31;
  int tile = blockIdx.x * 4 + wave;
  if (tile >= rowTiles) return;
  int row0 = tile * 16;
  int ml = lane & 15, hf = lane >> 4;
  const __bf16* arow = A + (size_t)(row0 + ml) * K;
  AFrag a[4];
  for (int kc = 0; kc < kChunks; ++kc) load_a_chunk(arow, kc, hf, a[kc].u);
#pragma unroll
  for (int nt = 0; nt < 4; ++nt) {
    v8f acc = splat8(bias[nt * 16 + ml]);
    for (int kc = 0; kc < kChunks; ++kc)
      acc = wmma_bf16(a[kc].v, load_b(Bp, nt, kc, kChunks, lane), acc);
#pragma unroll
    for (int r = 0; r < 8; ++r)
      Y[(size_t)(row0 + hf * 8 + r) * HIDN + nt * 16 + ml] = acc[r];
  }
}

// -------- batchnorm stats (column sums / sumsq via atomics) --------
__global__ void col_stats_kernel(const float* __restrict__ Y, float* __restrict__ stats, int rows) {
  int t = threadIdx.x;
  int c = t & 63;
  float s = 0.f, q = 0.f;
  for (int r = blockIdx.x * (blockDim.x >> 6) + (t >> 6); r < rows; r += gridDim.x * (blockDim.x >> 6)) {
    float v = Y[(size_t)r * HIDN + c];
    s += v;
    q += v * v;
  }
  atomicAdd(&stats[c], s);
  atomicAdd(&stats[64 + c], q);
}

__global__ void bn_apply_kernel(const float* __restrict__ Y, const float* __restrict__ stats,
                                const float* __restrict__ g, const float* __restrict__ b,
                                float* __restrict__ h, __bf16* __restrict__ hbf, int n) {
  int i = blockIdx.x * blockDim.x + threadIdx.x;
  if (i >= n) return;
  int c = i & 63;
  const float invN = 1.0f / (float)NNODES;
  float m = stats[c] * invN;
  float var = stats[64 + c] * invN - m * m;
  float rs = rsqrtf(var + EPSV);
  float v = (Y[i] - m) * rs * g[c] + b[c];
  v = fmaxf(v, 0.f);
  h[i] = v;
  hbf[i] = (__bf16)v;
}

// -------- LSTM path-conv: one wave per 16-edge tile, WMMA bf16, scatter to agg --------
// Packed W_ih followed by W_hh (64 KB total) staged into LDS with ONE TDM DMA.
struct WaveLds {
  int dsts[16];
  __bf16 h1[16 * HIDN]; // 2 KB
};

#define BIH_ELEMS (16 * 2 * 512)      // 16384 bf16 = 32 KB (one matrix)
#define BW_ELEMS  (2 * BIH_ELEMS)     // both matrices: 32768 bf16 = 64 KB

__global__ __launch_bounds__(128) void lstm_conv_kernel(const __bf16* __restrict__ hbf,
                                                        const int* __restrict__ paths,
                                                        const __bf16* __restrict__ Bpair,
                                                        const float* __restrict__ bs,
                                                        float* __restrict__ agg) {
  __shared__ __attribute__((aligned(16))) __bf16 sBW[BW_ELEMS]; // [Bih | Bhh]
  __shared__ WaveLds smem[4];

  // ---- stage both packed weight matrices into LDS via one Tensor Data Mover op ----
  if (threadIdx.x < 32) {
    uint32_t ldsOff = (uint32_t)(uintptr_t)(void*)&sBW[0];
    tdm_load_lds_1d(ldsOff, Bpair, BW_ELEMS);
    __builtin_amdgcn_s_wait_tensorcnt(0);
  }
  __syncthreads();
  const __bf16* sBih = sBW;
  const __bf16* sBhh = sBW + BIH_ELEMS;

  int wave = threadIdx.x >> 5;
  int lane = threadIdx.x & 31;
  int tile = blockIdx.x * 4 + wave; // 12500 * 4 == 50000 exactly
  int e0 = tile * 16;
  int ml = lane & 15, hf = lane >> 4;
  WaveLds& wl = smem[wave];

  int src = paths[(size_t)(e0 + ml) * 2 + 0];
  int dst = paths[(size_t)(e0 + ml) * 2 + 1];
  if (hf == 0) wl.dsts[ml] = dst;

  const __bf16* srow = hbf + (size_t)src * HIDN;
  const __bf16* drow = hbf + (size_t)dst * HIDN;
  AFrag ax0[2], ax1[2];
#pragma unroll
  for (int kc = 0; kc < 2; ++kc) {
    load_a_chunk(srow, kc, hf, ax0[kc].u);
    load_a_chunk(drow, kc, hf, ax1[kc].u);
  }

  v8f c1[4];
  // ---- step 1 (h0 = c0 = 0; f-gate irrelevant) ----
#pragma unroll
  for (int jt = 0; jt < 4; ++jt) {
    int ni = jt, ng = 8 + jt, no = 12 + jt;
    v8f gi = splat8(bs[ni * 16 + ml]);
    v8f gg = splat8(bs[ng * 16 + ml]);
    v8f go = splat8(bs[no * 16 + ml]);
#pragma unroll
    for (int kc = 0; kc < 2; ++kc) {
      gi = wmma_bf16(ax0[kc].v, load_b(sBih, ni, kc, 2, lane), gi);
      gg = wmma_bf16(ax0[kc].v, load_b(sBih, ng, kc, 2, lane), gg);
      go = wmma_bf16(ax0[kc].v, load_b(sBih, no, kc, 2, lane), go);
    }
#pragma unroll
    for (int r = 0; r < 8; ++r) {
      float i1 = sigf(gi[r]);
      float g1 = tanhf(gg[r]);
      float o1 = sigf(go[r]);
      float cc = i1 * g1;
      c1[jt][r] = cc;
      float h1v = o1 * tanhf(cc);
      int mm = hf * 8 + r;
      wl.h1[mm * HIDN + jt * 16 + ml] = (__bf16)h1v;
    }
  }
  __syncthreads();

  // ---- reload h1 in A layout ----
  AFrag ah[2];
  const __bf16* hrow = wl.h1 + (size_t)ml * HIDN;
#pragma unroll
  for (int kc = 0; kc < 2; ++kc) {
#pragma unroll
    for (int j = 0; j < 8; ++j) {
      int kk = kc * 32 + a_kbase(j, hf);
      ah[kc].u[j] = *(const u32*)(hrow + kk);
    }
  }

  // ---- step 2 ----
#pragma unroll
  for (int jt = 0; jt < 4; ++jt) {
    int ni = jt, nf = 4 + jt, ng = 8 + jt, no = 12 + jt;
    v8f gi = splat8(bs[ni * 16 + ml]);
    v8f gf = splat8(bs[nf * 16 + ml]);
    v8f gg = splat8(bs[ng * 16 + ml]);
    v8f go = splat8(bs[no * 16 + ml]);
#pragma unroll
    for (int kc = 0; kc < 2; ++kc) {
      gi = wmma_bf16(ax1[kc].v, load_b(sBih, ni, kc, 2, lane), gi);
      gi = wmma_bf16(ah[kc].v, load_b(sBhh, ni, kc, 2, lane), gi);
      gf = wmma_bf16(ax1[kc].v, load_b(sBih, nf, kc, 2, lane), gf);
      gf = wmma_bf16(ah[kc].v, load_b(sBhh, nf, kc, 2, lane), gf);
      gg = wmma_bf16(ax1[kc].v, load_b(sBih, ng, kc, 2, lane), gg);
      gg = wmma_bf16(ah[kc].v, load_b(sBhh, ng, kc, 2, lane), gg);
      go = wmma_bf16(ax1[kc].v, load_b(sBih, no, kc, 2, lane), go);
      go = wmma_bf16(ah[kc].v, load_b(sBhh, no, kc, 2, lane), go);
    }
#pragma unroll
    for (int r = 0; r < 8; ++r) {
      float i2 = sigf(gi[r]);
      float f2 = sigf(gf[r]);
      float g2 = tanhf(gg[r]);
      float o2 = sigf(go[r]);
      float c2 = f2 * c1[jt][r] + i2 * g2;
      float h2 = o2 * tanhf(c2);
      int mm = hf * 8 + r;
      int d = wl.dsts[mm];
      atomicAdd(&agg[(size_t)d * HIDN + jt * 16 + ml], h2);
    }
  }
}

// -------- residual + layernorm + relu (wave per row) --------
__global__ __launch_bounds__(128) void residual_ln_kernel(const float* __restrict__ agg,
                                                          float* __restrict__ h, __bf16* __restrict__ hbf,
                                                          const float* __restrict__ g,
                                                          const float* __restrict__ b, int rows) {
  int wave = threadIdx.x >> 5;
  int lane = threadIdx.x & 31;
  int row = blockIdx.x * 4 + wave;
  if (row >= rows) return;
  size_t base = (size_t)row * HIDN;
  float a0 = agg[base + lane] + h[base + lane];
  float a1 = agg[base + lane + 32] + h[base + lane + 32];
  float s = a0 + a1, q = a0 * a0 + a1 * a1;
  for (int o = 16; o > 0; o >>= 1) {
    s += __shfl_xor(s, o, 32);
    q += __shfl_xor(q, o, 32);
  }
  float m = s * (1.f / 64.f);
  float var = q * (1.f / 64.f) - m * m;
  float rs = rsqrtf(var + EPSV);
  float v0 = fmaxf((a0 - m) * rs * g[lane] + b[lane], 0.f);
  float v1 = fmaxf((a1 - m) * rs * g[lane + 32] + b[lane + 32], 0.f);
  h[base + lane] = v0;
  h[base + lane + 32] = v1;
  hbf[base + lane] = (__bf16)v0;
  hbf[base + lane + 32] = (__bf16)v1;
}

__global__ void pool_sum_kernel(const float* __restrict__ h, float* __restrict__ pooled, int rows) {
  int t = threadIdx.x;
  int c = t & 63;
  float s = 0.f;
  for (int r = blockIdx.x * (blockDim.x >> 6) + (t >> 6); r < rows; r += gridDim.x * (blockDim.x >> 6))
    s += h[(size_t)r * HIDN + c];
  atomicAdd(&pooled[c], s);
}

// -------- head: LN(pooled) -> logits -> softmax -> survival cumprod --------
__global__ void head_kernel(const float* __restrict__ pooled, const float* __restrict__ gg,
                            const float* __restrict__ gb, const float* __restrict__ Wout,
                            const float* __restrict__ bout, float* __restrict__ out) {
  int lane = threadIdx.x; // 32 threads
  float p0 = pooled[lane], p1 = pooled[lane + 32];
  float s = p0 + p1, q = p0 * p0 + p1 * p1;
  for (int o = 16; o > 0; o >>= 1) {
    s += __shfl_xor(s, o, 32);
    q += __shfl_xor(q, o, 32);
  }
  float m = s * (1.f / 64.f);
  float var = q * (1.f / 64.f) - m * m;
  float rs = rsqrtf(var + EPSV);
  float h0 = (p0 - m) * rs * gg[lane] + gb[lane];
  float h1 = (p1 - m) * rs * gg[lane + 32] + gb[lane + 32];
  out[lane] = h0;
  out[lane + 32] = h1;
  float lg[NCLS];
  for (int k = 0; k < NCLS; ++k) {
    float pt = h0 * Wout[k * HIDN + lane] + h1 * Wout[k * HIDN + lane + 32];
    for (int o = 16; o > 0; o >>= 1) pt += __shfl_xor(pt, o, 32);
    lg[k] = pt + bout[k];
  }
  if (lane == 0) {
    float mx = lg[0];
    for (int k = 1; k < NCLS; ++k) mx = fmaxf(mx, lg[k]);
    float ex[NCLS], se = 0.f;
    for (int k = 0; k < NCLS; ++k) {
      ex[k] = __expf(lg[k] - mx);
      se += ex[k];
    }
    float cp = 1.f;
    for (int k = 0; k < NCLS; ++k) {
      float pr = ex[k] / se;
      out[64 + k] = pr;
      cp *= (1.f - pr);
      out[74 + k] = cp;
    }
  }
}

// ======================= host launcher =======================
extern "C" void kernel_launch(void* const* d_in, const int* in_sizes, int n_in,
                              void* d_out, int out_size, void* d_ws, size_t ws_size,
                              hipStream_t stream) {
  const float* x      = (const float*)d_in[0];
  const int*   paths  = (const int*)d_in[1];
  const float* W_enc1 = (const float*)d_in[2];
  const float* b_enc1 = (const float*)d_in[3];
  const float* g_bn1  = (const float*)d_in[4];
  const float* be_bn1 = (const float*)d_in[5];
  const float* W_enc2 = (const float*)d_in[6];
  const float* b_enc2 = (const float*)d_in[7];
  const float* g_bn2  = (const float*)d_in[8];
  const float* be_bn2 = (const float*)d_in[9];
  const float* W_ih   = (const float*)d_in[10];
  const float* W_hh   = (const float*)d_in[11];
  const float* b_ih   = (const float*)d_in[12];
  const float* b_hh   = (const float*)d_in[13];
  const float* ln_g   = (const float*)d_in[14];
  const float* ln_b   = (const float*)d_in[15];
  const float* gn_g   = (const float*)d_in[16];
  const float* gn_b   = (const float*)d_in[17];
  const float* W_out  = (const float*)d_in[18];
  const float* b_out  = (const float*)d_in[19];
  float* out = (float*)d_out;

  char* ws = (char*)d_ws;
  size_t off = 0;
  auto take = [&](size_t bytes) {
    size_t p = off;
    off = (off + bytes + 255) & ~(size_t)255;
    return p;
  };
  __bf16* xbf    = (__bf16*)(ws + take((size_t)NNODES * DIN * 2));
  float*  h      = (float*)(ws + take((size_t)NNODES * HIDN * 4));
  __bf16* hbf    = (__bf16*)(ws + take((size_t)NNODES * HIDN * 2));
  float*  y      = (float*)(ws + take((size_t)NNODES * HIDN * 4));
  float*  agg    = (float*)(ws + take((size_t)NNODES * HIDN * 4));
  float*  stats  = (float*)(ws + take(128 * 4));
  float*  pooled = (float*)(ws + take(64 * 4));
  __bf16* Benc1  = (__bf16*)(ws + take((size_t)4 * 4 * 512 * 2));
  __bf16* Benc2  = (__bf16*)(ws + take((size_t)4 * 2 * 512 * 2));
  __bf16* Bpair  = (__bf16*)(ws + take((size_t)BW_ELEMS * 2)); // [Bih | Bhh], contiguous 64 KB
  float*  bsum   = (float*)(ws + take(256 * 4));

  // ---- weight packing / conversions ----
  pack_b_kernel<<<(4 * 4 * 512 + 255) / 256, 256, 0, stream>>>(W_enc1, Benc1, DIN, 4, 4);
  pack_b_kernel<<<(4 * 2 * 512 + 255) / 256, 256, 0, stream>>>(W_enc2, Benc2, HIDN, 2, 4);
  pack_b_kernel<<<(16 * 2 * 512 + 255) / 256, 256, 0, stream>>>(W_ih, Bpair, HIDN, 2, 16);
  pack_b_kernel<<<(16 * 2 * 512 + 255) / 256, 256, 0, stream>>>(W_hh, Bpair + BIH_ELEMS, HIDN, 2, 16);
  bias_sum_kernel<<<1, 256, 0, stream>>>(b_ih, b_hh, bsum);
  f32_to_bf16_kernel<<<(NNODES * DIN + 255) / 256, 256, 0, stream>>>(x, xbf, NNODES * DIN);

  const int rowTiles = NNODES / 16;           // 3125
  const int gemmBlocks = (rowTiles + 3) / 4;  // 782
  const int elemN = NNODES * HIDN;            // 3.2M

  // ---- encoder 1: GEMM -> BN stats -> BN apply + relu ----
  (void)hipMemsetAsync(stats, 0, 128 * 4, stream);
  gemm_bf16_kernel<<<gemmBlocks, 128, 0, stream>>>(xbf, DIN, 4, Benc1, b_enc1, y, rowTiles);
  col_stats_kernel<<<256, 256, 0, stream>>>(y, stats, NNODES);
  bn_apply_kernel<<<(elemN + 255) / 256, 256, 0, stream>>>(y, stats, g_bn1, be_bn1, h, hbf, elemN);

  // ---- encoder 2 ----
  (void)hipMemsetAsync(stats, 0, 128 * 4, stream);
  gemm_bf16_kernel<<<gemmBlocks, 128, 0, stream>>>(hbf, HIDN, 2, Benc2, b_enc2, y, rowTiles);
  col_stats_kernel<<<256, 256, 0, stream>>>(y, stats, NNODES);
  bn_apply_kernel<<<(elemN + 255) / 256, 256, 0, stream>>>(y, stats, g_bn2, be_bn2, h, hbf, elemN);

  // ---- 2 path-conv rounds: LSTM over 2-node paths + scatter + residual LN ----
  for (int i = 0; i < 2; ++i) {
    (void)hipMemsetAsync(agg, 0, (size_t)elemN * 4, stream);
    lstm_conv_kernel<<<NEDGES / 64, 128, 0, stream>>>(hbf, paths, Bpair, bsum, agg);
    residual_ln_kernel<<<(NNODES + 3) / 4, 128, 0, stream>>>(agg, h, hbf, ln_g + i * HIDN,
                                                             ln_b + i * HIDN, NNODES);
  }

  // ---- pool + head ----
  (void)hipMemsetAsync(pooled, 0, 64 * 4, stream);
  pool_sum_kernel<<<256, 256, 0, stream>>>(h, pooled, NNODES);
  head_kernel<<<1, 32, 0, stream>>>(pooled, gn_g, gn_b, W_out, b_out, out);
}
